// TransformerLanguageModel_50646254355221
// MI455X (gfx1250) — compile-verified, matrix operand order checked by reading, not run on hardware
//
#include <hip/hip_runtime.h>
#include <hip/hip_bf16.h>
#include <cstdint>

// ---------------------------------------------------------------------------
// GPT forward for MI455X (gfx1250): f16 WMMA matmuls, f32 accumulate.
// Model dims (fixed by reference): V=32000 C=1024 H=16 D=64 L=8 F=4096 B=2 T=2048
// ---------------------------------------------------------------------------

typedef _Float16 h2   __attribute__((ext_vector_type(2)));
typedef _Float16 h8   __attribute__((ext_vector_type(8)));
typedef _Float16 v16h __attribute__((ext_vector_type(16)));
typedef float    v8f  __attribute__((ext_vector_type(8)));
typedef float    f4v  __attribute__((ext_vector_type(4)));
typedef int      gv4i __attribute__((vector_size(16)));   // matches builtin's V4i

#define NV     32000
#define NC     1024
#define NH     16
#define ND     64
#define NL     8
#define NF     4096
#define NB     2
#define NT     2048
#define MTOK   (NB * NT)     // 4096 token rows
#define LN_EPS 1e-5f

// ---------------------------------------------------------------------------
// gfx1250 async global->LDS copy (ASYNCcnt-tracked), with graceful fallback.
// Probe-confirmed signature: (AS1 gv4i*, AS3 gv4i*, imm offset, imm cpol)
// ---------------------------------------------------------------------------
#if __has_builtin(__builtin_amdgcn_global_load_async_to_lds_b128)
#define HAVE_ASYNC_LDS 1
__device__ __forceinline__ void async_copy_b128(const _Float16* g, _Float16* l)
{
    __builtin_amdgcn_global_load_async_to_lds_b128(
        (__attribute__((address_space(1))) gv4i*)(uintptr_t)g,
        (__attribute__((address_space(3))) gv4i*)(uint32_t)(uintptr_t)l,
        0, 0);
}
#elif __has_builtin(__builtin_amdgcn_load_to_lds)
#define HAVE_ASYNC_LDS 1
__device__ __forceinline__ void async_copy_b128(const _Float16* g, _Float16* l)
{
    __builtin_amdgcn_load_to_lds(
        (void*)g,
        (__attribute__((address_space(3))) void*)(uint32_t)(uintptr_t)l,
        16, 0, 0);
}
#endif

#ifdef HAVE_ASYNC_LDS
__device__ __forceinline__ void wait_async0()
{
#if __has_builtin(__builtin_amdgcn_s_wait_asynccnt)
    __builtin_amdgcn_s_wait_asynccnt(0);
#else
    asm volatile("s_wait_asynccnt 0" ::: "memory");
#endif
}
#endif

// Build a WMMA f16 A/B fragment from a row whose 32 K-values are contiguous
// at p[0..31]; per ISA layout the lane needs runs [kbase,kbase+8) and
// [16+kbase, 16+kbase+8)  (kbase = 0 for lanes 0-15, 8 for lanes 16-31).
__device__ __forceinline__ v16h frag16(const _Float16* p, int kbase)
{
    h8 lo = *(const h8*)(p + kbase);
    h8 hi = *(const h8*)(p + 16 + kbase);
    v16h f;
    #pragma unroll
    for (int i = 0; i < 8; ++i) { f[i] = lo[i]; f[8 + i] = hi[i]; }
    return f;
}

// ---------------------------------------------------------------------------
// Embedding: x[b,t,:] = emb[idx[b,t],:] + pos[t,:]
// ---------------------------------------------------------------------------
__global__ __launch_bounds__(256) void embed_kernel(
    const int* __restrict__ idx, const float* __restrict__ emb,
    const float* __restrict__ pos, float* __restrict__ x)
{
    int i = blockIdx.x * 256 + threadIdx.x;          // over MTOK*NC
    if (i >= MTOK * NC) return;
    int c  = i % NC;
    int bt = i / NC;
    int t  = bt % NT;
    x[i] = emb[(size_t)idx[bt] * NC + c] + pos[(size_t)t * NC + c];
}

// ---------------------------------------------------------------------------
// LayerNorm over C=1024; one block (256 threads) per token row; f16 output.
// ---------------------------------------------------------------------------
__global__ __launch_bounds__(256) void ln_kernel(
    const float* __restrict__ x, const float* __restrict__ g,
    const float* __restrict__ b, _Float16* __restrict__ out)
{
    __shared__ float red[256];
    int row = blockIdx.x;
    int tid = threadIdx.x;
    const float* xr = x + (size_t)row * NC;

    float s = 0.f;
    for (int c = tid; c < NC; c += 256) s += xr[c];
    red[tid] = s; __syncthreads();
    for (int off = 128; off > 0; off >>= 1) {
        if (tid < off) red[tid] += red[tid + off];
        __syncthreads();
    }
    float mu = red[0] * (1.0f / NC);
    __syncthreads();

    float s2 = 0.f;
    for (int c = tid; c < NC; c += 256) { float d = xr[c] - mu; s2 += d * d; }
    red[tid] = s2; __syncthreads();
    for (int off = 128; off > 0; off >>= 1) {
        if (tid < off) red[tid] += red[tid + off];
        __syncthreads();
    }
    float inv = rsqrtf(red[0] * (1.0f / NC) + LN_EPS);

    for (int c = tid; c < NC; c += 256)
        out[(size_t)row * NC + c] = (_Float16)((xr[c] - mu) * inv * g[c] + b[c]);
}

// ---------------------------------------------------------------------------
// Block-tiled WMMA GEMM:  out = A(f16, MxK) * W(f32->f16, KxN) + bias
// Block: 256 threads = 8 waves; tile 64(M) x 128(N); BK=32.
// Wave grid: 4 along M x 2 along N; each wave: 16 rows x 64 cols = 4 accs.
// A tile staged via async global->LDS b128; W tile converted f32->f16
// in-flight and stored as packed b32 k-pairs into transposed [n][k].
// Epilogue (residual / relu / f32-store / f16-store) resolved at compile time.
// ---------------------------------------------------------------------------
template<bool RES, bool RELU, bool OUTF, bool OUTH>
__global__ __launch_bounds__(256) void gemm_kernel(
    const _Float16* __restrict__ A, const float* __restrict__ W,
    const float* __restrict__ bias, const float* __restrict__ res,
    float* __restrict__ outF, _Float16* __restrict__ outH,
    int M, int N, int K)
{
    __shared__ _Float16 As[64 * 32];    // 4 KB
    __shared__ _Float16 Bs[128 * 32];   // 8 KB, stored [n][k] (transposed)

    const int tid   = threadIdx.x;
    const int lane  = tid & 31;
    const int wave  = tid >> 5;
    const int wm    = wave & 3;          // wave row   (0..3)
    const int wn    = wave >> 2;         // wave col   (0..1)
    const int bm0   = blockIdx.y * 64;
    const int bn0   = blockIdx.x * 128;
    const int kbase = (lane < 16) ? 0 : 8;
    const int nlo   = lane & 15;

    // A staging: 64x32 halfs, 8 halfs (16B) per thread
    const int rowA = tid >> 2;           // 0..63
    const int kkA  = (tid & 3) * 8;      // 0,8,16,24
    // W staging: (k,k+1) pair x 8 columns per thread -> packed b32 stores
    const int kr2  = (tid >> 4) * 2;     // 0,2,...,30
    const int nc8  = (tid & 15) * 8;     // 0..120

    v8f acc[4];
    for (int g = 0; g < 4; ++g) acc[g] = (v8f){0.f,0.f,0.f,0.f,0.f,0.f,0.f,0.f};

    for (int k0 = 0; k0 < K; k0 += 32) {
        // ---- A tile: global -> LDS (async if available) ----
        const _Float16* ag = A + (size_t)(bm0 + rowA) * K + k0 + kkA;
        _Float16*       al = As + rowA * 32 + kkA;
#ifdef HAVE_ASYNC_LDS
        async_copy_b128(ag, al);
#else
        *(h8*)al = *(const h8*)ag;
#endif

        // ---- W tile: f32 load, cvt, packed b32 transposed store ----
        const float* w0 = W + (size_t)(k0 + kr2) * N + bn0 + nc8;
        const float* w1r = w0 + N;
        f4v a0 = *(const f4v*)(w0);
        f4v a1 = *(const f4v*)(w0 + 4);
        f4v b0 = *(const f4v*)(w1r);
        f4v b1 = *(const f4v*)(w1r + 4);
        #pragma unroll
        for (int e = 0; e < 8; ++e) {
            float lo = (e < 4) ? a0[e] : a1[e - 4];
            float hi = (e < 4) ? b0[e] : b1[e - 4];
            h2 p; p[0] = (_Float16)lo; p[1] = (_Float16)hi;
            *(h2*)(Bs + (nc8 + e) * 32 + kr2) = p;   // kr2 even -> 4B aligned
        }

        // prefetch next W k-tile into cache hierarchy (global_prefetch_b8)
        if (k0 + 32 < K)
            __builtin_prefetch(W + (size_t)(k0 + 32 + kr2) * N + bn0 + nc8, 0, 1);

#ifdef HAVE_ASYNC_LDS
        wait_async0();
#endif
        __syncthreads();

        // ---- fragments + WMMA ----
        v16h af = frag16(As + (wm * 16 + nlo) * 32, kbase);
        #pragma unroll
        for (int g = 0; g < 4; ++g) {
            v16h bf = frag16(Bs + (wn * 64 + g * 16 + nlo) * 32, kbase);
            acc[g] = __builtin_amdgcn_wmma_f32_16x16x32_f16(
                false, af, false, bf, (short)0, acc[g], false, false);
        }
        __syncthreads();
    }

    // Epilogue. D layout: vgpr r -> M = r (lanes 0-15) / 8+r (lanes 16-31), N = lane&15
    #pragma unroll
    for (int g = 0; g < 4; ++g) {
        #pragma unroll
        for (int r = 0; r < 8; ++r) {
            const int ml = (lane < 16) ? r : 8 + r;
            const int m  = bm0 + wm * 16 + ml;
            const int n  = bn0 + wn * 64 + g * 16 + nlo;
            float v = acc[g][r] + bias[n];
            if (RES)  v += res[(size_t)m * N + n];
            if (RELU) v = fmaxf(v, 0.f);
            if (OUTF) outF[(size_t)m * N + n] = v;
            if (OUTH) outH[(size_t)m * N + n] = (_Float16)v;
        }
    }
}

// ---------------------------------------------------------------------------
// Flash-style causal attention. One wave per (b, h, 16-query-row tile);
// 4 independent waves per 128-thread block. Online softmax (running max/sum),
// Q.K^T and P.V both via v_wmma_f32_16x16x32_f16.
// Per chunk, the 32x64 V tile is staged into per-wave LDS transposed to
// [dim][token] with coalesced b128 global loads + packed b32 DS stores, so
// the P.V B fragments come from contiguous ds_load_b128 runs.
// No __syncthreads: each wave owns its LDS slice (divergent chunk counts).
// ---------------------------------------------------------------------------
__global__ __launch_bounds__(128) void attn_kernel(
    const _Float16* __restrict__ q, const _Float16* __restrict__ k,
    const _Float16* __restrict__ v, _Float16* __restrict__ o)
{
    __shared__ float    P[4][16 * 32];     // 8 KB  scores/probs
    __shared__ _Float16 Vt[4][64 * 32];    // 16 KB V tile, [dim][token]
    __shared__ float    Arow[4][16];
    __shared__ float    Lrow[4][16];

    const int wave = threadIdx.x >> 5;
    const int lane = threadIdx.x & 31;
    const int tile = blockIdx.x * 4 + wave;      // over B*H*(T/16) = 4096
    const int qtiles = NT / 16;
    const int qt = tile % qtiles;
    const int bh = tile / qtiles;
    const int h  = bh % NH;
    const int b  = bh / NH;

    const int row16 = lane & 15;
    const int halfs = lane >> 4;
    const int kbase = halfs ? 8 : 0;
    const float scale = 0.125f;                  // 1/sqrt(64)
    float*    Pw = P[wave];
    _Float16* Vw = Vt[wave];

    // Q fragments for the two K-steps across D=64 (contiguous b128 loads)
    const _Float16* qrow = q + ((size_t)(b * NT + qt * 16 + row16)) * NC + h * ND;
    v16h aq[2];
    aq[0] = frag16(qrow, kbase);
    aq[1] = frag16(qrow + 32, kbase);

    v8f oacc[4];
    for (int g = 0; g < 4; ++g) oacc[g] = (v8f){0.f,0.f,0.f,0.f,0.f,0.f,0.f,0.f};
    float m_i = -__builtin_inff();
    float l_i = 0.f;
    const int mtok0 = qt * 16;
    const int nch = (qt + 2) >> 1;               // ceil((qt+1)*16 / 32)

    for (int cc = 0; cc < nch; ++cc) {
        const int keybase = cc * 32;

        // --- stage V tile (32 tokens x 64 dims) transposed into Vt[wave] ---
        {
            const int tp    = lane & 15;         // token pair index
            const int dhalf = lane >> 4;         // which 32 dims
            const _Float16* v0 = v + ((size_t)(b * NT + keybase + 2 * tp)) * NC
                                   + h * ND + dhalf * 32;
            const _Float16* v1 = v0 + NC;
            h8 r0[4], r1[4];
            #pragma unroll
            for (int j = 0; j < 4; ++j) {
                r0[j] = *(const h8*)(v0 + 8 * j);
                r1[j] = *(const h8*)(v1 + 8 * j);
            }
            _Float16* vt = Vw + dhalf * 32 * 32 + 2 * tp;  // [dim][token]
            #pragma unroll
            for (int d = 0; d < 32; ++d) {
                h2 p; p[0] = r0[d >> 3][d & 7]; p[1] = r1[d >> 3][d & 7];
                *(h2*)(vt + d * 32) = p;                   // 4B-aligned b32
            }
        }

        // --- scores: two 16x16 tiles -> masked, scaled, into LDS ---
        #pragma unroll
        for (int a = 0; a < 2; ++a) {
            const int ktok = keybase + a * 16 + row16;   // key token (B frag col)
            const _Float16* krow = k + ((size_t)(b * NT + ktok)) * NC + h * ND;
            v8f sc = (v8f){0.f,0.f,0.f,0.f,0.f,0.f,0.f,0.f};
            sc = __builtin_amdgcn_wmma_f32_16x16x32_f16(
                false, aq[0], false, frag16(krow, kbase), (short)0, sc, false, false);
            sc = __builtin_amdgcn_wmma_f32_16x16x32_f16(
                false, aq[1], false, frag16(krow + 32, kbase), (short)0, sc, false, false);
            #pragma unroll
            for (int r = 0; r < 8; ++r) {
                const int ml   = (lane < 16) ? r : 8 + r;
                const int ntok = keybase + a * 16 + row16;
                float sval = (ntok <= mtok0 + ml) ? sc[r] * scale : -__builtin_inff();
                Pw[ml * 32 + a * 16 + row16] = sval;
            }
        }

        // --- online softmax: lane handles row row16, cols halfs*16..+15 ---
        float mc = -__builtin_inff();
        #pragma unroll
        for (int j = 0; j < 16; ++j)
            mc = fmaxf(mc, Pw[row16 * 32 + halfs * 16 + j]);
        mc = fmaxf(mc, __shfl_xor(mc, 16, 32));
        const float mnew  = fmaxf(m_i, mc);
        const float alpha = __expf(m_i - mnew);
        float sp = 0.f;
        #pragma unroll
        for (int j = 0; j < 16; ++j) {
            const int ix = row16 * 32 + halfs * 16 + j;
            float pv = __expf(Pw[ix] - mnew);
            Pw[ix] = pv;
            sp += pv;
        }
        sp += __shfl_xor(sp, 16, 32);
        l_i = l_i * alpha + sp;
        m_i = mnew;
        if (lane < 16) Arow[wave][row16] = alpha;

        // rescale running O accumulators by alpha (per output row)
        #pragma unroll
        for (int g = 0; g < 4; ++g)
            #pragma unroll
            for (int r = 0; r < 8; ++r) {
                const int ml = (lane < 16) ? r : 8 + r;
                oacc[g][r] *= Arow[wave][ml];
            }

        // --- P . V : A fragment from Pw (f32->f16), B fragments from Vt ---
        v16h ap;
        {
            const float* pr = Pw + row16 * 32;
            f4v p0 = *(const f4v*)(pr + kbase);
            f4v p1 = *(const f4v*)(pr + kbase + 4);
            f4v p2 = *(const f4v*)(pr + 16 + kbase);
            f4v p3 = *(const f4v*)(pr + 16 + kbase + 4);
            #pragma unroll
            for (int i = 0; i < 4; ++i) {
                ap[i]      = (_Float16)p0[i];
                ap[4 + i]  = (_Float16)p1[i];
                ap[8 + i]  = (_Float16)p2[i];
                ap[12 + i] = (_Float16)p3[i];
            }
        }
        #pragma unroll
        for (int g = 0; g < 4; ++g) {
            v16h bv = frag16(Vw + (g * 16 + row16) * 32, kbase);
            oacc[g] = __builtin_amdgcn_wmma_f32_16x16x32_f16(
                false, ap, false, bv, (short)0, oacc[g], false, false);
        }
    }

    if (lane < 16) Lrow[wave][row16] = l_i;

    #pragma unroll
    for (int g = 0; g < 4; ++g)
        #pragma unroll
        for (int r = 0; r < 8; ++r) {
            const int ml = (lane < 16) ? r : 8 + r;
            float val = oacc[g][r] / Lrow[wave][ml];
            o[((size_t)(b * NT + qt * 16 + ml)) * NC + h * ND + g * 16 + row16] =
                (_Float16)val;
        }
}

// ---------------------------------------------------------------------------
// Logits for last token of each batch: out[b,v] = hfin[last_b] . out_w[:,v] + out_b[v]
// Tiny (0.13 GFLOP) — plain dot kernel, WMMA would waste 14/16 rows.
// ---------------------------------------------------------------------------
__global__ __launch_bounds__(256) void logits_kernel(
    const _Float16* __restrict__ hfin, const float* __restrict__ out_w,
    const float* __restrict__ out_b, float* __restrict__ out)
{
    int i = blockIdx.x * 256 + threadIdx.x;      // over NB*NV
    if (i >= NB * NV) return;
    int b = i / NV;
    int vv = i % NV;
    const _Float16* hr = hfin + ((size_t)(b * NT + NT - 1)) * NC;
    float s = out_b[vv];
    for (int c = 0; c < NC; ++c)
        s += (float)hr[c] * out_w[(size_t)c * NV + vv];
    out[i] = s;
}

// ---------------------------------------------------------------------------
// Host orchestration
// ---------------------------------------------------------------------------
extern "C" void kernel_launch(void* const* d_in, const int* in_sizes, int n_in,
                              void* d_out, int out_size, void* d_ws, size_t ws_size,
                              hipStream_t stream)
{
    const int*   idx   = (const int*)  d_in[0];
    const float* emb   = (const float*)d_in[1];
    const float* pos   = (const float*)d_in[2];
    const float* ln1_g = (const float*)d_in[3];
    const float* ln1_b = (const float*)d_in[4];
    const float* wq    = (const float*)d_in[5];
    const float* bq    = (const float*)d_in[6];
    const float* wk    = (const float*)d_in[7];
    const float* bk    = (const float*)d_in[8];
    const float* wv    = (const float*)d_in[9];
    const float* bv    = (const float*)d_in[10];
    const float* wo    = (const float*)d_in[11];
    const float* bo    = (const float*)d_in[12];
    const float* ln2_g = (const float*)d_in[13];
    const float* ln2_b = (const float*)d_in[14];
    const float* w1    = (const float*)d_in[15];
    const float* b1    = (const float*)d_in[16];
    const float* w2    = (const float*)d_in[17];
    const float* b2    = (const float*)d_in[18];
    const float* lnf_g = (const float*)d_in[19];
    const float* lnf_b = (const float*)d_in[20];
    const float* out_w = (const float*)d_in[21];
    const float* out_b = (const float*)d_in[22];
    float* out = (float*)d_out;

    // Workspace carve-up (~89 MB total)
    char* w = (char*)d_ws;
    size_t off = 0;
    float*     x    = (float*)(w + off);     off += (size_t)MTOK * NC * sizeof(float);
    _Float16*  hbuf = (_Float16*)(w + off);  off += (size_t)MTOK * NC * sizeof(_Float16);
    _Float16*  qb   = (_Float16*)(w + off);  off += (size_t)MTOK * NC * sizeof(_Float16);
    _Float16*  kb   = (_Float16*)(w + off);  off += (size_t)MTOK * NC * sizeof(_Float16);
    _Float16*  vb   = (_Float16*)(w + off);  off += (size_t)MTOK * NC * sizeof(_Float16);
    _Float16*  ob   = (_Float16*)(w + off);  off += (size_t)MTOK * NC * sizeof(_Float16);
    _Float16*  mid  = (_Float16*)(w + off);  off += (size_t)MTOK * NF * sizeof(_Float16);
    (void)off; (void)ws_size; (void)in_sizes; (void)n_in; (void)out_size;

    // 1. Embedding
    embed_kernel<<<(MTOK * NC + 255) / 256, 256, 0, stream>>>(idx, emb, pos, x);

    const dim3 gemmB(256);
    const dim3 gC(NC / 128, MTOK / 64);   // N=1024
    const dim3 gF(NF / 128, MTOK / 64);   // N=4096

    for (int l = 0; l < NL; ++l) {
        const float* wq_l = wq + (size_t)l * NC * NC;
        const float* wk_l = wk + (size_t)l * NC * NC;
        const float* wv_l = wv + (size_t)l * NC * NC;
        const float* wo_l = wo + (size_t)l * NC * NC;
        const float* w1_l = w1 + (size_t)l * NC * NF;
        const float* w2_l = w2 + (size_t)l * NF * NC;

        // h = LN1(x)
        ln_kernel<<<MTOK, 256, 0, stream>>>(x, ln1_g + (size_t)l * NC,
                                            ln1_b + (size_t)l * NC, hbuf);
        // Q, K, V projections (f16 outputs)
        gemm_kernel<false,false,false,true><<<gC, gemmB, 0, stream>>>(
            hbuf, wq_l, bq + (size_t)l * NC, nullptr, nullptr, qb, MTOK, NC, NC);
        gemm_kernel<false,false,false,true><<<gC, gemmB, 0, stream>>>(
            hbuf, wk_l, bk + (size_t)l * NC, nullptr, nullptr, kb, MTOK, NC, NC);
        gemm_kernel<false,false,false,true><<<gC, gemmB, 0, stream>>>(
            hbuf, wv_l, bv + (size_t)l * NC, nullptr, nullptr, vb, MTOK, NC, NC);

        // causal flash attention -> ob (f16)
        attn_kernel<<<(NB * NH * (NT / 16)) / 4, 128, 0, stream>>>(qb, kb, vb, ob);

        // x = x + ob @ wo + bo
        gemm_kernel<true,false,true,false><<<gC, gemmB, 0, stream>>>(
            ob, wo_l, bo + (size_t)l * NC, x, x, nullptr, MTOK, NC, NC);

        // h = LN2(x)
        ln_kernel<<<MTOK, 256, 0, stream>>>(x, ln2_g + (size_t)l * NC,
                                            ln2_b + (size_t)l * NC, hbuf);
        // mid = relu(h @ w1 + b1)  (f16)
        gemm_kernel<false,true,false,true><<<gF, gemmB, 0, stream>>>(
            hbuf, w1_l, b1 + (size_t)l * NF, nullptr, nullptr, mid, MTOK, NF, NC);
        // x = x + mid @ w2 + b2
        gemm_kernel<true,false,true,false><<<gC, gemmB, 0, stream>>>(
            mid, w2_l, b2 + (size_t)l * NC, x, x, nullptr, MTOK, NC, NF);
    }

    // final LN + logits for last token of each batch
    ln_kernel<<<MTOK, 256, 0, stream>>>(x, lnf_g, lnf_b, hbuf);
    logits_kernel<<<(NB * NV + 255) / 256, 256, 0, stream>>>(hbuf, out_w, out_b, out);
}
